// StageModule_89318139888128
// MI455X (gfx1250) — matrix operand order repaired and checked by reading, not compiled, not run on hardware
//
#include <hip/hip_runtime.h>
#include <math.h>

// ---------------------------------------------------------------------------
// CDNA5 (gfx1250) Swin-3D block: all matmuls via v_wmma_f32_16x16x32_bf16.
// ---------------------------------------------------------------------------

typedef __attribute__((ext_vector_type(16))) __bf16 v16bf;
typedef __attribute__((ext_vector_type(8)))  float  v8f;
typedef __attribute__((ext_vector_type(8)))  __bf16 bf16x8;

static constexpr int BATCH = 2;
static constexpr int GS    = 32;              // grid after patch merge
static constexpr int CH    = 192;
static constexpr int NTOK  = BATCH * GS * GS * GS;   // 65536
static constexpr int NWIN  = BATCH * 512;            // 1024 windows
static constexpr int NQKV  = 576;
static constexpr int HIDD  = 768;

// ---------------------------------------------------------------------------
// Weight pack: fp32 [K][N] -> bf16 [N][K]  (so B-fragments are contiguous)
// ---------------------------------------------------------------------------
__global__ void pack_wt_kernel(const float* __restrict__ src,
                               unsigned short* __restrict__ dst_, int K, int N) {
  __bf16* dst = (__bf16*)dst_;
  int total = K * N;
  for (int idx = blockIdx.x * blockDim.x + threadIdx.x; idx < total;
       idx += gridDim.x * blockDim.x) {
    int k = idx % K;
    int n = idx / K;
    dst[idx] = (__bf16)src[(size_t)k * N + n];
  }
}

// ---------------------------------------------------------------------------
// Patch merging: gather 768 = (c,dd,dh,dw) values per token, LayerNorm(768),
// write bf16 rows (GEMM A operand). Token order = natural (b,z,y,x).
// ---------------------------------------------------------------------------
__global__ __launch_bounds__(256)
void patch_merge_ln_kernel(const float* __restrict__ x,
                           const float* __restrict__ g,
                           const float* __restrict__ b,
                           unsigned short* __restrict__ out_) {
  __bf16* out = (__bf16*)out_;
  int t   = blockIdx.x;
  int tid = threadIdx.x;
  int bb  = t >> 15;
  int rem = t & 32767;
  int z = rem >> 10, y = (rem >> 5) & 31, xx = rem & 31;

  float v[3], s = 0.f, s2 = 0.f;
  for (int i = 0; i < 3; ++i) {
    int j = tid + 256 * i;
    int c = j >> 3, dd = (j >> 2) & 1, dh = (j >> 1) & 1, dw = j & 1;
    size_t src = ((((size_t)bb * 64 + (2 * z + dd)) * 64 + (2 * y + dh)) * 64 +
                  (2 * xx + dw)) * 96 + c;
    v[i] = x[src];
    s  += v[i];
    s2 += v[i] * v[i];
  }
  __shared__ float sm[256], sq[256];
  sm[tid] = s; sq[tid] = s2; __syncthreads();
  for (int o = 128; o > 0; o >>= 1) {
    if (tid < o) { sm[tid] += sm[tid + o]; sq[tid] += sq[tid + o]; }
    __syncthreads();
  }
  float mean = sm[0] * (1.f / 768.f);
  float var  = sq[0] * (1.f / 768.f) - mean * mean;
  float inv  = rsqrtf(var + 1e-5f);
  for (int i = 0; i < 3; ++i) {
    int j = tid + 256 * i;
    out[(size_t)t * 768 + j] = (__bf16)((v[i] - mean) * inv * g[j] + b[j]);
  }
}

// ---------------------------------------------------------------------------
// Generic bf16 WMMA GEMM: C[M][N] = A[M][K] @ B(packed [N][K]).
// Block tile 128x64, 8 waves (4x2) of 2x2 16x16 tiles, K-step 32.
// Epilogue variants via template.
// ---------------------------------------------------------------------------
enum { EPI_BIAS_F32 = 0, EPI_BF16 = 1, EPI_BIAS_GELU_BF16 = 2, EPI_BIAS_RES_F32 = 3 };

template <int EPI>
__global__ __launch_bounds__(256)
void gemm_bf16_kernel(const unsigned short* __restrict__ A_,
                      const unsigned short* __restrict__ Bp_,
                      const float* __restrict__ bias,
                      const float* __restrict__ resid,
                      float* __restrict__ outF,
                      unsigned short* __restrict__ outB_,
                      int M, int N, int K) {
  const __bf16* A  = (const __bf16*)A_;
  const __bf16* Bp = (const __bf16*)Bp_;
  __bf16* outB = (__bf16*)outB_;

  __shared__ __bf16 As[128 * 32];
  __shared__ __bf16 Bs[64 * 32];

  int tid  = threadIdx.x;
  int lane = tid & 31;
  int wave = tid >> 5;
  int wm = wave & 3, wn = wave >> 2;        // wave grid 4x2 -> 128x64 tile
  int hi = lane >> 4, lr = lane & 15;
  int row0 = blockIdx.y * 128;
  int col0 = blockIdx.x * 64;

  v8f acc[2][2];
  for (int a = 0; a < 2; ++a)
    for (int b = 0; b < 2; ++b)
      for (int i = 0; i < 8; ++i) acc[a][b][i] = 0.f;

  for (int k0 = 0; k0 < K; k0 += 32) {
    // stage A (128x32) and B (64x32) in LDS with b128 loads
    {
      int q = tid;
      int r = q >> 2, kc = (q & 3) * 8;
      *(bf16x8*)&As[r * 32 + kc] =
          *(const bf16x8*)&A[(size_t)(row0 + r) * K + k0 + kc];
      q = tid + 256;
      r = q >> 2; kc = (q & 3) * 8;
      *(bf16x8*)&As[r * 32 + kc] =
          *(const bf16x8*)&A[(size_t)(row0 + r) * K + k0 + kc];
      int n = tid >> 2; kc = (tid & 3) * 8;
      *(bf16x8*)&Bs[n * 32 + kc] =
          *(const bf16x8*)&Bp[(size_t)(col0 + n) * K + k0 + kc];
      if (k0 + 32 < K)   // hint next K-slab into cache (global_prefetch_b8)
        __builtin_prefetch(&A[(size_t)(row0 + (tid >> 2)) * K + k0 + 32], 0, 1);
    }
    __syncthreads();

    // fragments in contiguous k order, k = hi*16 + i (A 16x32 bf16, B 32x16 bf16)
    v16bf af[2], bf[2];
    for (int tr = 0; tr < 2; ++tr) {
      int rrow = wm * 32 + tr * 16 + lr;
      const __bf16* p0 = &As[rrow * 32 + hi * 16];       // K = hi*16 + i
      const __bf16* p1 = &As[rrow * 32 + hi * 16 + 8];   // K = hi*16 + 8 + i
      for (int i = 0; i < 8; ++i) { af[tr][i] = p0[i]; af[tr][8 + i] = p1[i]; }
    }
    for (int tc = 0; tc < 2; ++tc) {
      int n = wn * 32 + tc * 16 + lr;                    // column = lane
      const __bf16* p = &Bs[n * 32 + hi * 16];           // K = hi*16 + i
      for (int i = 0; i < 16; ++i) bf[tc][i] = p[i];
    }
    for (int tr = 0; tr < 2; ++tr)
      for (int tc = 0; tc < 2; ++tc)
        acc[tr][tc] = __builtin_amdgcn_wmma_f32_16x16x32_bf16(
            false, af[tr], false, bf[tc], (short)0, acc[tr][tc], false, false);
    __syncthreads();
  }

  // epilogue (C layout: lanes 0-15 M=r, lanes 16-31 M=8+r; N = lane&15)
  for (int tr = 0; tr < 2; ++tr)
    for (int tc = 0; tc < 2; ++tc)
      for (int r = 0; r < 8; ++r) {
        int m = row0 + wm * 32 + tr * 16 + (hi ? 8 + r : r);
        int n = col0 + wn * 32 + tc * 16 + lr;
        float v = acc[tr][tc][r];
        size_t idx = (size_t)m * N + n;
        if constexpr (EPI == EPI_BIAS_F32) {
          outF[idx] = v + bias[n];
        } else if constexpr (EPI == EPI_BF16) {
          outB[idx] = (__bf16)v;
        } else if constexpr (EPI == EPI_BIAS_GELU_BF16) {
          float gx = v + bias[n];
          outB[idx] = (__bf16)(0.5f * gx *
                               (1.f + erff(gx * 0.70710678118654752440f)));
        } else {  // EPI_BIAS_RES_F32
          outF[idx] = v + bias[n] + resid[idx];
        }
      }
}

// ---------------------------------------------------------------------------
// "Weird view" window partition (faithful to torch .view of (B,D,H,W,C)):
// xw[t][c] = Xnat_flat[b*6291456 + c*32768 + Dz*4096 + wz*1024
//                      + Hy*128 + wy*32 + Wx*4 + wx]
// ---------------------------------------------------------------------------
__global__ void partition_kernel(const float* __restrict__ in,
                                 float* __restrict__ out) {
  size_t total = (size_t)NTOK * CH;
  for (size_t idx = (size_t)blockIdx.x * blockDim.x + threadIdx.x; idx < total;
       idx += (size_t)gridDim.x * blockDim.x) {
    int t = (int)(idx / CH), c = (int)(idx % CH);
    int W = t >> 6, n = t & 63;
    int bb = W >> 9, Dz = (W >> 6) & 7, Hy = (W >> 3) & 7, Wx = W & 7;
    int wz = n >> 4, wy = (n >> 2) & 3, wx = n & 3;
    size_t f = (size_t)bb * 6291456 + (size_t)c * 32768 + Dz * 4096 +
               wz * 1024 + Hy * 128 + wy * 32 + Wx * 4 + wx;
    out[idx] = in[f];
  }
}

// ---------------------------------------------------------------------------
// LayerNorm(192) over window-ordered residual stream -> bf16 (GEMM A operand)
// ---------------------------------------------------------------------------
__global__ __launch_bounds__(64)
void ln192_kernel(const float* __restrict__ X, const float* __restrict__ g,
                  const float* __restrict__ b, unsigned short* __restrict__ out_) {
  __bf16* out = (__bf16*)out_;
  int t = blockIdx.x, tid = threadIdx.x;
  float v[3], s = 0.f, s2 = 0.f;
  for (int i = 0; i < 3; ++i) {
    int c = tid + 64 * i;
    v[i] = X[(size_t)t * CH + c];
    s += v[i]; s2 += v[i] * v[i];
  }
  __shared__ float sm[64], sq[64];
  sm[tid] = s; sq[tid] = s2; __syncthreads();
  for (int o = 32; o > 0; o >>= 1) {
    if (tid < o) { sm[tid] += sm[tid + o]; sq[tid] += sq[tid + o]; }
    __syncthreads();
  }
  float mean = sm[0] * (1.f / 192.f);
  float var  = sq[0] * (1.f / 192.f) - mean * mean;
  float inv  = rsqrtf(var + 1e-5f);
  for (int i = 0; i < 3; ++i) {
    int c = tid + 64 * i;
    out[(size_t)t * CH + c] = (__bf16)((v[i] - mean) * inv * g[c] + b[c]);
  }
}

// ---------------------------------------------------------------------------
// Fused: window_reverse -> roll(-2,-2,-2) -> weird-view partition -> LN(192)
// (one pure index composition -> gather from X, then LayerNorm)
// ---------------------------------------------------------------------------
__global__ __launch_bounds__(64)
void ln_shift_kernel(const float* __restrict__ X, const float* __restrict__ g,
                     const float* __restrict__ b, unsigned short* __restrict__ out_) {
  __bf16* out = (__bf16*)out_;
  int t = blockIdx.x, tid = threadIdx.x;
  int W = t >> 6, n = t & 63;
  int bb = W >> 9, Dz = (W >> 6) & 7, Hy = (W >> 3) & 7, Wx = W & 7;
  int wz = n >> 4, wy = (n >> 2) & 3, wx = n & 3;
  int base = Dz * 4096 + wz * 1024 + Hy * 128 + wy * 32 + Wx * 4 + wx;

  float v[3], s = 0.f, s2 = 0.f;
  for (int i = 0; i < 3; ++i) {
    int c = tid + 64 * i;
    int f = c * 32768 + base;                 // weird-view flat (per batch)
    int d = f / 196608;  int r  = f - d * 196608;
    int h = r / 6144;    int r2 = r - h * 6144;
    int w = r2 / 192;    int cc = r2 - w * 192;
    int d2 = (d + 2) & 31, h2 = (h + 2) & 31, w2 = (w + 2) & 31;  // roll(-2)
    int W2 = ((bb * 8 + (d2 >> 2)) * 8 + (h2 >> 2)) * 8 + (w2 >> 2);
    int n2 = ((d2 & 3) * 4 + (h2 & 3)) * 4 + (w2 & 3);
    v[i] = X[((size_t)W2 * 64 + n2) * CH + cc];
    s += v[i]; s2 += v[i] * v[i];
  }
  __shared__ float sm[64], sq[64];
  sm[tid] = s; sq[tid] = s2; __syncthreads();
  for (int o = 32; o > 0; o >>= 1) {
    if (tid < o) { sm[tid] += sm[tid + o]; sq[tid] += sq[tid + o]; }
    __syncthreads();
  }
  float mean = sm[0] * (1.f / 192.f);
  float var  = sq[0] * (1.f / 192.f) - mean * mean;
  float inv  = rsqrtf(var + 1e-5f);
  for (int i = 0; i < 3; ++i) {
    int c = tid + 64 * i;
    out[(size_t)t * CH + c] = (__bf16)((v[i] - mean) * inv * g[c] + b[c]);
  }
}

// ---------------------------------------------------------------------------
// Window attention: one wave per (window, head).
//   S = scale*(q@k^T) + rel_bias [+ shift mask]; softmax; O = P@V.
//   16 + 16 WMMAs per wave, all operands staged in LDS.
// ---------------------------------------------------------------------------
template <bool MASKED>
__global__ __launch_bounds__(32)
void attention_kernel(const unsigned short* __restrict__ QKV_,
                      const float* __restrict__ bt,
                      unsigned short* __restrict__ out_) {
  const __bf16* QKV = (const __bf16*)QKV_;
  __bf16* out = (__bf16*)out_;

  __shared__ __bf16 qs[64 * 32];
  __shared__ __bf16 ks[64 * 32];
  __shared__ __bf16 vs[64 * 32];
  __shared__ float  Sf[64 * 64];
  __shared__ __bf16 Pb[64 * 64];
  __shared__ float  biasH[343];

  int h = blockIdx.x;
  int W = blockIdx.y;
  int lane = threadIdx.x;
  int hi = lane >> 4, lr = lane & 15;
  size_t t0 = (size_t)W * 64;
  int qoff = h * 32;

  for (int i = lane; i < 2048; i += 32) {
    int r = i >> 5, c = i & 31;
    size_t rowb = (t0 + r) * NQKV + qoff + c;
    qs[i] = QKV[rowb];
    ks[i] = QKV[rowb + 192];
    vs[i] = QKV[rowb + 384];
  }
  for (int j = lane; j < 343; j += 32) biasH[j] = bt[j * 6 + h];
  __syncthreads();

  int widx = W & 511;  // window id within batch for the shift mask
  int dzw = (widx >> 6) & 7, hyw = (widx >> 3) & 7, wxw = widx & 7;
  const float scale = 0.17677669529663687f;  // 32^-0.5

  for (int ti = 0; ti < 4; ++ti) {
    v16bf a;
    {
      int row = ti * 16 + lr;
      const __bf16* p0 = &qs[row * 32 + hi * 16];
      const __bf16* p1 = &qs[row * 32 + hi * 16 + 8];
      for (int i = 0; i < 8; ++i) { a[i] = p0[i]; a[8 + i] = p1[i]; }
    }
    for (int tj = 0; tj < 4; ++tj) {
      v16bf bk;
      {
        int col = tj * 16 + lr;  // key token = B column; B = k^T so row of k
        const __bf16* p = &ks[col * 32 + hi * 16];
        for (int i = 0; i < 16; ++i) bk[i] = p[i];
      }
      v8f c;
      for (int i = 0; i < 8; ++i) c[i] = 0.f;
      c = __builtin_amdgcn_wmma_f32_16x16x32_bf16(false, a, false, bk,
                                                  (short)0, c, false, false);
      for (int r = 0; r < 8; ++r) {
        int m  = ti * 16 + (hi ? 8 + r : r);   // query index
        int nn = tj * 16 + lr;                 // key index
        int zm = m >> 4, ym = (m >> 2) & 3, xm = m & 3;
        int zn = nn >> 4, yn = (nn >> 2) & 3, xn = nn & 3;
        int rel = (zm - zn + 3) * 49 + (ym - yn + 3) * 7 + (xm - xn + 3);
        float val = c[r] * scale + biasH[rel];
        if (MASKED) {
          auto rid = [](int vv) { return vv < 28 ? 0 : (vv < 30 ? 1 : 2); };
          int rm = rid(dzw * 4 + zm) * 9 + rid(hyw * 4 + ym) * 3 + rid(wxw * 4 + xm);
          int rn = rid(dzw * 4 + zn) * 9 + rid(hyw * 4 + yn) * 3 + rid(wxw * 4 + xn);
          if (rm != rn) val -= 100.f;
        }
        Sf[m * 64 + nn] = val;
      }
    }
  }
  __syncthreads();

  // softmax over keys (2 rows per lane), P in bf16
  for (int rr = 0; rr < 2; ++rr) {
    int r = lane + rr * 32;
    float mx = -1e30f;
    for (int j = 0; j < 64; ++j) mx = fmaxf(mx, Sf[r * 64 + j]);
    float sum = 0.f;
    for (int j = 0; j < 64; ++j) {
      float e = __expf(Sf[r * 64 + j] - mx);
      Sf[r * 64 + j] = e;
      sum += e;
    }
    float inv = 1.f / sum;
    for (int j = 0; j < 64; ++j) Pb[r * 64 + j] = (__bf16)(Sf[r * 64 + j] * inv);
  }
  __syncthreads();

  // O = P(64x64) @ V(64x32)
  for (int ti = 0; ti < 4; ++ti) {
    for (int tj = 0; tj < 2; ++tj) {
      v8f c;
      for (int i = 0; i < 8; ++i) c[i] = 0.f;
      for (int kst = 0; kst < 2; ++kst) {
        v16bf a, bv;
        {
          int row = ti * 16 + lr;
          const __bf16* p0 = &Pb[row * 64 + kst * 32 + hi * 16];
          const __bf16* p1 = &Pb[row * 64 + kst * 32 + hi * 16 + 8];
          for (int i = 0; i < 8; ++i) { a[i] = p0[i]; a[8 + i] = p1[i]; }
        }
        {
          int col = tj * 16 + lr;
          for (int i = 0; i < 16; ++i) {
            int kk = kst * 32 + hi * 16 + i;
            bv[i] = vs[kk * 32 + col];
          }
        }
        c = __builtin_amdgcn_wmma_f32_16x16x32_bf16(false, a, false, bv,
                                                    (short)0, c, false, false);
      }
      for (int r = 0; r < 8; ++r) {
        int m = ti * 16 + (hi ? 8 + r : r);
        int d = tj * 16 + lr;
        out[(t0 + m) * CH + h * 32 + d] = (__bf16)c[r];
      }
    }
  }
}

// ---------------------------------------------------------------------------
// Final standard window_reverse: (B*512,64,C) -> (B,32,32,32,C) fp32 output
// ---------------------------------------------------------------------------
__global__ void reverse_out_kernel(const float* __restrict__ X,
                                   float* __restrict__ out) {
  size_t total = (size_t)NTOK * CH;
  for (size_t idx = (size_t)blockIdx.x * blockDim.x + threadIdx.x; idx < total;
       idx += (size_t)gridDim.x * blockDim.x) {
    int c = (int)(idx % CH);
    size_t rest = idx / CH;
    int w = (int)(rest % 32); rest /= 32;
    int h = (int)(rest % 32); rest /= 32;
    int d = (int)(rest % 32);
    int bb = (int)(rest / 32);
    int W = ((bb * 8 + (d >> 2)) * 8 + (h >> 2)) * 8 + (w >> 2);
    int n = ((d & 3) * 4 + (h & 3)) * 4 + (w & 3);
    out[idx] = X[((size_t)W * 64 + n) * CH + c];
  }
}

// ---------------------------------------------------------------------------
// Host orchestration
// ---------------------------------------------------------------------------
extern "C" void kernel_launch(void* const* d_in, const int* in_sizes, int n_in,
                              void* d_out, int out_size, void* d_ws, size_t ws_size,
                              hipStream_t stream) {
  const float* x      = (const float*)d_in[0];
  const float* pm_g   = (const float*)d_in[1];
  const float* pm_b   = (const float*)d_in[2];
  const float* pm_w   = (const float*)d_in[3];
  const float* pm_lb  = (const float*)d_in[4];
  const float* n1_g   = (const float*)d_in[5];
  const float* n1_b   = (const float*)d_in[6];
  const float* n2_g   = (const float*)d_in[7];
  const float* n2_b   = (const float*)d_in[8];
  const float* a1_qkv = (const float*)d_in[9];
  const float* a1_pw  = (const float*)d_in[10];
  const float* a1_pb  = (const float*)d_in[11];
  const float* a1_bt  = (const float*)d_in[12];
  const float* a2_qkv = (const float*)d_in[13];
  const float* a2_pw  = (const float*)d_in[14];
  const float* a2_pb  = (const float*)d_in[15];
  const float* a2_bt  = (const float*)d_in[16];
  const float* m1_g   = (const float*)d_in[17];
  const float* m1_b   = (const float*)d_in[18];
  const float* m1_w1  = (const float*)d_in[19];
  const float* m1_b1  = (const float*)d_in[20];
  const float* m1_w2  = (const float*)d_in[21];
  const float* m1_b2  = (const float*)d_in[22];
  const float* m2_g   = (const float*)d_in[23];
  const float* m2_b   = (const float*)d_in[24];
  const float* m2_w1  = (const float*)d_in[25];
  const float* m2_b1  = (const float*)d_in[26];
  const float* m2_w2  = (const float*)d_in[27];
  const float* m2_b2  = (const float*)d_in[28];
  float* outp = (float*)d_out;

  char* w = (char*)d_ws;
  size_t o = 0;
  auto alloc = [&](size_t bytes) {
    size_t r = o;
    o += (bytes + 255) & ~(size_t)255;
    return r;
  };
  size_t oWpm = alloc((size_t)192 * 768 * 2);
  size_t oWq1 = alloc((size_t)576 * 192 * 2);
  size_t oWp1 = alloc((size_t)192 * 192 * 2);
  size_t oWq2 = alloc((size_t)576 * 192 * 2);
  size_t oWp2 = alloc((size_t)192 * 192 * 2);
  size_t oW11 = alloc((size_t)768 * 192 * 2);
  size_t oW12 = alloc((size_t)192 * 768 * 2);
  size_t oW21 = alloc((size_t)768 * 192 * 2);
  size_t oW22 = alloc((size_t)192 * 768 * 2);
  size_t oXnat = alloc((size_t)NTOK * CH * 4);   // natural-order fp32
  size_t oX    = alloc((size_t)NTOK * CH * 4);   // window-order fp32 residual
  size_t oAb   = alloc((size_t)NTOK * 768 * 2);  // bf16 A operand (768-wide)
  size_t oQKV  = alloc((size_t)NTOK * NQKV * 2); // bf16 qkv
  size_t oLN   = alloc((size_t)NTOK * CH * 2);   // bf16 layernormed acts
  size_t oOb   = alloc((size_t)NTOK * CH * 2);   // bf16 attention output
  (void)ws_size; (void)n_in; (void)in_sizes; (void)out_size;

  unsigned short* Wpm = (unsigned short*)(w + oWpm);
  unsigned short* Wq1 = (unsigned short*)(w + oWq1);
  unsigned short* Wp1 = (unsigned short*)(w + oWp1);
  unsigned short* Wq2 = (unsigned short*)(w + oWq2);
  unsigned short* Wp2 = (unsigned short*)(w + oWp2);
  unsigned short* W11 = (unsigned short*)(w + oW11);
  unsigned short* W12 = (unsigned short*)(w + oW12);
  unsigned short* W21 = (unsigned short*)(w + oW21);
  unsigned short* W22 = (unsigned short*)(w + oW22);
  float* Xnat = (float*)(w + oXnat);
  float* X    = (float*)(w + oX);
  unsigned short* Ab  = (unsigned short*)(w + oAb);
  unsigned short* QKV = (unsigned short*)(w + oQKV);
  unsigned short* LN  = (unsigned short*)(w + oLN);
  unsigned short* Ob  = (unsigned short*)(w + oOb);

  dim3 b256(256), b64(64), b32(32);
  dim3 gPack(256);
  // ---- weight packs: src [K][N] fp32 -> dst [N][K] bf16 ----
  pack_wt_kernel<<<gPack, b256, 0, stream>>>(pm_w,   Wpm, 768, 192);
  pack_wt_kernel<<<gPack, b256, 0, stream>>>(a1_qkv, Wq1, 192, 576);
  pack_wt_kernel<<<gPack, b256, 0, stream>>>(a1_pw,  Wp1, 192, 192);
  pack_wt_kernel<<<gPack, b256, 0, stream>>>(a2_qkv, Wq2, 192, 576);
  pack_wt_kernel<<<gPack, b256, 0, stream>>>(a2_pw,  Wp2, 192, 192);
  pack_wt_kernel<<<gPack, b256, 0, stream>>>(m1_w1,  W11, 192, 768);
  pack_wt_kernel<<<gPack, b256, 0, stream>>>(m1_w2,  W12, 768, 192);
  pack_wt_kernel<<<gPack, b256, 0, stream>>>(m2_w1,  W21, 192, 768);
  pack_wt_kernel<<<gPack, b256, 0, stream>>>(m2_w2,  W22, 768, 192);

  dim3 gTok(NTOK);
  dim3 gG3(3, NTOK / 128), gG9(9, NTOK / 128), gG12(12, NTOK / 128);
  dim3 gElem(4096);
  dim3 gAttn(6, NWIN);

  // ---- patch merging: LN(768) + GEMM 768->192 (+pm_lb) -> Xnat fp32 ----
  patch_merge_ln_kernel<<<gTok, b256, 0, stream>>>(x, pm_g, pm_b, Ab);
  gemm_bf16_kernel<EPI_BIAS_F32><<<gG3, b256, 0, stream>>>(
      Ab, Wpm, pm_lb, nullptr, Xnat, nullptr, NTOK, 192, 768);
  partition_kernel<<<gElem, b256, 0, stream>>>(Xnat, X);

  // ---- block 1: window attention (no mask) + FFN ----
  ln192_kernel<<<gTok, b64, 0, stream>>>(X, n1_g, n1_b, LN);
  gemm_bf16_kernel<EPI_BF16><<<gG9, b256, 0, stream>>>(
      LN, Wq1, nullptr, nullptr, nullptr, QKV, NTOK, NQKV, 192);
  attention_kernel<false><<<gAttn, b32, 0, stream>>>(QKV, a1_bt, Ob);
  gemm_bf16_kernel<EPI_BIAS_RES_F32><<<gG3, b256, 0, stream>>>(
      Ob, Wp1, a1_pb, X, X, nullptr, NTOK, 192, 192);
  ln192_kernel<<<gTok, b64, 0, stream>>>(X, m1_g, m1_b, LN);
  gemm_bf16_kernel<EPI_BIAS_GELU_BF16><<<gG12, b256, 0, stream>>>(
      LN, W11, m1_b1, nullptr, nullptr, Ab, NTOK, HIDD, 192);
  gemm_bf16_kernel<EPI_BIAS_RES_F32><<<gG3, b256, 0, stream>>>(
      Ab, W12, m1_b2, X, X, nullptr, NTOK, 192, HIDD);

  // ---- block 2: shifted window attention (mask) + FFN ----
  ln_shift_kernel<<<gTok, b64, 0, stream>>>(X, n2_g, n2_b, LN);
  gemm_bf16_kernel<EPI_BF16><<<gG9, b256, 0, stream>>>(
      LN, Wq2, nullptr, nullptr, nullptr, QKV, NTOK, NQKV, 192);
  attention_kernel<true><<<gAttn, b32, 0, stream>>>(QKV, a2_bt, Ob);
  gemm_bf16_kernel<EPI_BIAS_RES_F32><<<gG3, b256, 0, stream>>>(
      Ob, Wp2, a2_pb, X, X, nullptr, NTOK, 192, 192);
  ln192_kernel<<<gTok, b64, 0, stream>>>(X, m2_g, m2_b, LN);
  gemm_bf16_kernel<EPI_BIAS_GELU_BF16><<<gG12, b256, 0, stream>>>(
      LN, W21, m2_b1, nullptr, nullptr, Ab, NTOK, HIDD, 192);
  gemm_bf16_kernel<EPI_BIAS_RES_F32><<<gG3, b256, 0, stream>>>(
      Ab, W22, m2_b2, X, X, nullptr, NTOK, 192, HIDD);

  // ---- final standard window_reverse -> fp32 output ----
  reverse_out_kernel<<<gElem, b256, 0, stream>>>(X, outp);
}
